// CircleLoss_v2_58411555225706
// MI455X (gfx1250) — matrix operand order repaired and checked by reading, not compiled, not run on hardware
//
#include <hip/hip_runtime.h>
#include <hip/hip_bf16.h>
#include <math.h>

typedef _Float16 half_t;
typedef __attribute__((ext_vector_type(16))) _Float16 v16h;
typedef __attribute__((ext_vector_type(8)))  _Float16 v8h;
typedef __attribute__((ext_vector_type(8)))  float    v8f;
typedef __attribute__((__vector_size__(16))) int      vi4;   // matches builtin's V4i

#define D   128      // feature dim (fixed by reference)
#define BT  64       // block tile edge (rows/cols of sim per workgroup)
#define LDK 136      // padded LDS row stride in halfs (136*2B = 272B, 16B aligned)
#define NEG_BIG (-1.0e30f)

// ---- CDNA5 async global->LDS copy (ASYNCcnt-tracked), with safe fallback ----
#if defined(__gfx1250__) && __has_builtin(__builtin_amdgcn_global_load_async_to_lds_b128)
#define HAVE_ASYNC_LDS 1
__device__ __forceinline__ void async_copy_b128(const void* g, const void* l) {
    // generic->AS1: global addresses are identity in the generic aperture
    // generic->AS3: low 32 bits of a generic LDS address are the LDS offset (ISA 10.2)
    auto gp = (__attribute__((address_space(1))) vi4*)(unsigned long long)(uintptr_t)g;
    auto lp = (__attribute__((address_space(3))) vi4*)(unsigned)(uintptr_t)l;
    __builtin_amdgcn_global_load_async_to_lds_b128(gp, lp, 0, 0);
}
__device__ __forceinline__ void wait_async0() {
#if __has_builtin(__builtin_amdgcn_s_wait_asynccnt)
    __builtin_amdgcn_s_wait_asynccnt(0);
#else
    asm volatile("s_wait_asynccnt 0x0" ::: "memory");
#endif
}
#else
#define HAVE_ASYNC_LDS 0
#endif

// merge running logsumexp state (m,s) with partial (m2,s2): lse = m + log(s)
__device__ __forceinline__ void lse_merge(float& m, float& s, float m2, float s2) {
    float M = fmaxf(m, m2);
    s = s * __expf(m - M) + s2 * __expf(m2 - M);
    m = M;
}

// ---------------- Kernel 1: row L2-normalize, fp32 -> f16 ----------------
__global__ void __launch_bounds__(D)
nrm_kernel(const float* __restrict__ f, half_t* __restrict__ fn) {
    __shared__ float red[D];
    const int row = blockIdx.x;
    const int t   = threadIdx.x;
    float v = f[(size_t)row * D + t];
    red[t] = v * v;
    __syncthreads();
    for (int off = D / 2; off > 0; off >>= 1) {
        if (t < off) red[t] += red[t + off];
        __syncthreads();
    }
    float nrm = sqrtf(red[0]);
    float inv = 1.0f / fmaxf(nrm, 1e-12f);
    fn[(size_t)row * D + t] = (half_t)(v * inv);
}

// ---------------- Kernel 2: fused WMMA sim tiles + masked tile-LSE ----------------
__global__ void __launch_bounds__(256)
sim_tile_kernel(const half_t* __restrict__ fn, const int* __restrict__ label,
                float* __restrict__ partials, int NB) {
    __shared__ __align__(16) half_t Asub[BT][LDK];
    __shared__ __align__(16) half_t Bsub[BT][LDK];
    __shared__ int   labA[BT], labB[BT];
    __shared__ float rm[256], rs[256];

    // map linear block id -> upper-triangular block pair (bi <= bj)
    int p = blockIdx.x;
    int bi = 0;
    while (p >= NB - bi) { p -= NB - bi; ++bi; }
    const int bj = bi + p;
    const int ibase = bi * BT, jbase = bj * BT;

    const int t = threadIdx.x;

    // stage both row panels (BT x 128 halfs each) into LDS, 16B per lane per iter
    #pragma unroll
    for (int q = 0; q < (BT * D / 8) / 256; ++q) {
        int idx = q * 256 + t;          // 0 .. BT*16-1
        int r   = idx >> 4;             // row within tile
        int cc  = idx & 15;             // 8-half column chunk
        const half_t* gA = fn + (size_t)(ibase + r) * D + cc * 8;
        const half_t* gB = fn + (size_t)(jbase + r) * D + cc * 8;
#if HAVE_ASYNC_LDS
        async_copy_b128(gA, &Asub[r][cc * 8]);
        async_copy_b128(gB, &Bsub[r][cc * 8]);
#else
        *(uint4*)&Asub[r][cc * 8] = *(const uint4*)gA;
        *(uint4*)&Bsub[r][cc * 8] = *(const uint4*)gB;
#endif
    }
    if (t < BT)        labA[t]      = label[ibase + t];
    else if (t < 2*BT) labB[t - BT] = label[jbase + (t - BT)];
#if HAVE_ASYNC_LDS
    wait_async0();
#endif
    __syncthreads();

    const int lane = t & 31;
    const int wave = t >> 5;
    const int hM   = lane & 15;   // column within 16x16 tile (N index)
    const int hK   = lane >> 4;   // half-select per ISA fragment layout

    float m_p = NEG_BIG, s_p = 0.f, m_n = NEG_BIG, s_n = 0.f;

    // 16 wmma tiles (4x4) per block; wave w owns row ti = w>>1, tiles tj = (w&1)*2 + {0,1}
    const int ti = wave >> 1;

    // hoist A fragments (K = 128 -> 4 chunks of 32), reused for both tj tiles
    v16h afrag[D / 32];
    #pragma unroll
    for (int kc = 0; kc < D / 32; ++kc) {
        const int aRow = ti * 16 + hM;
        const int aOff = kc * 32 + hK * 8;
        v8h alo = *(const v8h*)&Asub[aRow][aOff];
        v8h ahi = *(const v8h*)&Asub[aRow][aOff + 16];
        afrag[kc] = __builtin_shufflevector(alo, ahi, 0,1,2,3,4,5,6,7,8,9,10,11,12,13,14,15);
    }

    #pragma unroll
    for (int mtl = 0; mtl < 2; ++mtl) {
        const int tj = (wave & 1) * 2 + mtl;
        if (bi == bj && tj < ti) continue;   // wave-uniform skip below diagonal

        v8f c = {0.f, 0.f, 0.f, 0.f, 0.f, 0.f, 0.f, 0.f};
        #pragma unroll
        for (int kc = 0; kc < D / 32; ++kc) {
            // B fragment: lane holds column n=hM, K = hK*8 .. +7 and 16 + hK*8 .. +7
            const int bRow = tj * 16 + hM;
            const int bOff = kc * 32 + hK * 8;
            v8h blo = *(const v8h*)&Bsub[bRow][bOff];
            v8h bhi = *(const v8h*)&Bsub[bRow][bOff + 16];
            v16h b = __builtin_shufflevector(blo, bhi, 0,1,2,3,4,5,6,7,8,9,10,11,12,13,14,15);
            c = __builtin_amdgcn_wmma_f32_16x16x32_f16(false, afrag[kc], false, b,
                                                       (short)0, c, false, false);
        }

        // ---- branchless masked epilogue with one deferred rescale per tile ----
        const int j  = jbase + tj * 16 + hM;
        const int lj = labB[tj * 16 + hM];
        float vp[8], vn[8];
        float tp = -INFINITY, tn = -INFINITY;
        #pragma unroll
        for (int r = 0; r < 8; ++r) {
            const int iloc = ti * 16 + hK * 8 + r;   // C layout: M = 8*hK + r
            const int i    = ibase + iloc;
            const float s  = c[r];
            const bool valid = (j > i);               // strict upper triangle
            const bool same  = (labA[iloc] == lj);
            const float lp = -fmaxf(1.25f - s, 0.f) * (s - 0.75f) * 256.f;
            const float ln =  fmaxf(s + 0.25f, 0.f) * (s - 0.25f) * 256.f;
            vp[r] = (valid && same)  ? lp : -INFINITY;
            vn[r] = (valid && !same) ? ln : -INFINITY;
            tp = fmaxf(tp, vp[r]);
            tn = fmaxf(tn, vn[r]);
        }
        // accumulator max has finite floor NEG_BIG, so exp args are never NaN
        const float Mp = fmaxf(m_p, tp);
        const float Mn = fmaxf(m_n, tn);
        s_p *= __expf(m_p - Mp);  m_p = Mp;
        s_n *= __expf(m_n - Mn);  m_n = Mn;
        #pragma unroll
        for (int r = 0; r < 8; ++r) {
            s_p += __expf(vp[r] - Mp);   // exp(-inf) = 0 for masked slots
            s_n += __expf(vn[r] - Mn);
        }
    }

    // block-level LSE reduction (pos then neg)
    rm[t] = m_p; rs[t] = s_p; __syncthreads();
    for (int off = 128; off > 0; off >>= 1) {
        if (t < off) { float m2 = rm[t+off], s2 = rs[t+off]; lse_merge(rm[t], rs[t], m2, s2); }
        __syncthreads();
    }
    if (t == 0) { partials[(size_t)blockIdx.x*4+0] = rm[0]; partials[(size_t)blockIdx.x*4+1] = rs[0]; }
    __syncthreads();
    rm[t] = m_n; rs[t] = s_n; __syncthreads();
    for (int off = 128; off > 0; off >>= 1) {
        if (t < off) { float m2 = rm[t+off], s2 = rs[t+off]; lse_merge(rm[t], rs[t], m2, s2); }
        __syncthreads();
    }
    if (t == 0) { partials[(size_t)blockIdx.x*4+2] = rm[0]; partials[(size_t)blockIdx.x*4+3] = rs[0]; }
}

// ---------------- Kernel 3: merge partials, softplus(lse_p + lse_n) ----------------
__global__ void __launch_bounds__(256)
finalize_kernel(const float* __restrict__ partials, int nBlk, float* __restrict__ out) {
    __shared__ float rm[256], rs[256], qm[256], qs[256];
    const int t = threadIdx.x;
    float m_p = NEG_BIG, s_p = 0.f, m_n = NEG_BIG, s_n = 0.f;
    for (int i = t; i < nBlk; i += 256) {
        lse_merge(m_p, s_p, partials[(size_t)i*4+0], partials[(size_t)i*4+1]);
        lse_merge(m_n, s_n, partials[(size_t)i*4+2], partials[(size_t)i*4+3]);
    }
    rm[t] = m_p; rs[t] = s_p; qm[t] = m_n; qs[t] = s_n;
    __syncthreads();
    for (int off = 128; off > 0; off >>= 1) {
        if (t < off) {
            lse_merge(rm[t], rs[t], rm[t+off], rs[t+off]);
            lse_merge(qm[t], qs[t], qm[t+off], qs[t+off]);
        }
        __syncthreads();
    }
    if (t == 0) {
        float lp = (rs[0] > 0.f) ? rm[0] + __logf(rs[0]) : NEG_BIG;
        float ln = (qs[0] > 0.f) ? qm[0] + __logf(qs[0]) : NEG_BIG;
        float x  = lp + ln;
        out[0] = (x > 20.f) ? x : log1pf(__expf(x));   // softplus, stable
    }
}

extern "C" void kernel_launch(void* const* d_in, const int* in_sizes, int n_in,
                              void* d_out, int out_size, void* d_ws, size_t ws_size,
                              hipStream_t stream) {
    (void)n_in; (void)out_size; (void)ws_size;
    const float* feature = (const float*)d_in[0];
    const int*   label   = (const int*)d_in[1];

    const int N      = in_sizes[0] / D;       // 8192
    const int NB     = N / BT;                // 128
    const int nPairs = NB * (NB + 1) / 2;     // 8256 upper-triangular block tiles

    half_t* fn       = (half_t*)d_ws;                                   // N*128 f16 = 2 MB
    float*  partials = (float*)((char*)d_ws + (size_t)N * D * sizeof(half_t));

    nrm_kernel<<<N, D, 0, stream>>>(feature, fn);
    sim_tile_kernel<<<nPairs, 256, 0, stream>>>(fn, label, partials, NB);
    finalize_kernel<<<1, 256, 0, stream>>>(partials, nPairs, (float*)d_out);
}
